// GAT_68667937128873
// MI455X (gfx1250) — matrix operand order repaired and checked by reading, not compiled, not run on hardware
//
#include <hip/hip_runtime.h>
#include <hip/hip_bf16.h>

// ---------------------------------------------------------------------------
// GAT forward for MI455X (gfx1250, wave32, WMMA).
// Dense GEMMs use v_wmma_f32_16x16x32_bf16 (bf16 inputs, f32 accumulate),
// one wave per 16x64 output strip (4 independent WMMA accumulator chains,
// A fragment reused 4x). Edge softmax/aggregation via f32 atomics with
// global_prefetch_b8 pre-warming of the xp[src] gather rows into L2.
// ---------------------------------------------------------------------------

#define HEADS 4
#define NEG_SLOPE 0.2f
#define LN_EPS 1e-5f

typedef __attribute__((ext_vector_type(16))) __bf16 v16bf;
typedef __attribute__((ext_vector_type(8)))  float  v8f;

// ---------------------------------------------------------------------------
// LayerNorm over 128 channels, one 128-thread block per node, bf16 output.
// ---------------------------------------------------------------------------
__global__ __launch_bounds__(128)
void ln_bf16_kernel(const float* __restrict__ X, const float* __restrict__ scale,
                    const float* __restrict__ bias, __bf16* __restrict__ H, int N) {
    __shared__ float red[128];
    const int n = blockIdx.x;
    const int t = threadIdx.x;
    if (n >= N) return;
    float v = X[(size_t)n * 128 + t];
    red[t] = v;
    __syncthreads();
    for (int s = 64; s > 0; s >>= 1) { if (t < s) red[t] += red[t + s]; __syncthreads(); }
    const float mu = red[0] * (1.0f / 128.0f);
    __syncthreads();
    const float dv = v - mu;
    red[t] = dv * dv;
    __syncthreads();
    for (int s = 64; s > 0; s >>= 1) { if (t < s) red[t] += red[t + s]; __syncthreads(); }
    const float var = red[0] * (1.0f / 128.0f);
    const float o = dv * rsqrtf(var + LN_EPS) * scale[t] + bias[t];
    H[(size_t)n * 128 + t] = (__bf16)o;
}

// ---------------------------------------------------------------------------
// Pack a row-major f32 weight W[K=128, P] into bf16 B-matrix fragments for
// v_wmma_f32_16x16x32_bf16.  Per (ktile,ntile) tile of 32x16: lane L (<16)
// holds column n=nt*16+L, K = kt*32 + 0..15; lane L (>=16) holds the same
// column, K = kt*32 + 16..31 (contiguous K order, same as the A loads).  Element index
// within a lane is contiguous in K, so the GEMM reads 2x b128 per lane.
// ---------------------------------------------------------------------------
__global__ void pack_b_kernel(const float* __restrict__ W, __bf16* __restrict__ Bp,
                              int K, int P) {
    const int idx = blockIdx.x * blockDim.x + threadIdx.x;
    if (idx >= K * P) return;
    const int ntiles = P >> 4;
    const int tile = idx >> 9;          // 512 bf16 per 32x16 tile
    const int rem  = idx & 511;
    const int lane = rem >> 4;          // 0..31
    const int pos  = rem & 15;          // K offset within lane
    const int kt = tile / ntiles;
    const int nt = tile % ntiles;
    const int nn = nt * 16 + (lane & 15);
    const int kk = kt * 32 + ((lane < 16) ? 0 : 16) + pos;
    Bp[idx] = (__bf16)W[(size_t)kk * P + nn];
}

// ---------------------------------------------------------------------------
// WMMA GEMM: Y[M,P] = A[M,128]_bf16 @ Bpacked + (bias).  One wave per 16x64
// output strip (4 adjacent 16x16 N-tiles): each A fragment (two b128 loads
// per lane) feeds 4 independent v_wmma_f32_16x16x32_bf16 chains, K=128
// unrolled as 4 k-steps -> 16 WMMAs per wave.
// A fragment (contiguous K order, as in packed B): lane L, row M = mt*16 + L%16,
//   dwords0-3 = K k0+base..k0+base+7, dwords4-7 = K k0+base+8.. (base = L<16?0:16)
// ---------------------------------------------------------------------------
__global__ __launch_bounds__(128)
void gemm_wmma_kernel(const __bf16* __restrict__ A, const __bf16* __restrict__ Bp,
                      const float* __restrict__ bias, float* __restrict__ Y,
                      int M, int P, int useBias) {
    const int lane = threadIdx.x & 31;
    const int wave = threadIdx.x >> 5;
    const int ntiles  = P >> 4;            // 16-wide tiles
    const int nstrips = P >> 6;            // 64-wide strips (4 tiles)
    const int mtiles  = (M + 15) >> 4;
    const int strip = blockIdx.x * 4 + wave;
    if (strip >= mtiles * nstrips) return;   // wave-uniform exit
    const int mt = strip / nstrips;
    const int ns = strip % nstrips;

    int arow = mt * 16 + (lane & 15);
    if (arow >= M) arow = M - 1;             // clamp loads; stores guarded below
    const __bf16* __restrict__ Arow = A + (size_t)arow * 128;
    const int base8 = (lane < 16) ? 0 : 16;

    v8f acc[4] = {v8f{}, v8f{}, v8f{}, v8f{}};
#pragma unroll
    for (int kt = 0; kt < 4; ++kt) {
        const int k0 = kt * 32;
        union { uint4 u[2]; v16bf v; } a;
        a.u[0] = *(const uint4*)(Arow + k0 + base8);
        a.u[1] = *(const uint4*)(Arow + k0 + 8 + base8);
#pragma unroll
        for (int j = 0; j < 4; ++j) {
            union { uint4 u[2]; v16bf v; } b;
            const __bf16* bt =
                Bp + ((size_t)(kt * ntiles + ns * 4 + j) << 9) + lane * 16;
            b.u[0] = *(const uint4*)(bt);
            b.u[1] = *(const uint4*)(bt + 8);
            // (neg_a, A, neg_b, B, c_mod, C, reuse_a, reuse_b)
            acc[j] = __builtin_amdgcn_wmma_f32_16x16x32_bf16(
                false, a.v, false, b.v, (short)0, acc[j], false, false);
        }
    }

    const int rbase = mt * 16 + ((lane < 16) ? 0 : 8);
#pragma unroll
    for (int j = 0; j < 4; ++j) {
        const int col = (ns * 4 + j) * 16 + (lane & 15);
        const float bv = useBias ? bias[col] : 0.0f;
#pragma unroll
        for (int r = 0; r < 8; ++r) {
            const int row = rbase + r;
            if (row < M) Y[(size_t)row * P + col] = acc[j][r] + bv;
        }
    }
}

// ---------------------------------------------------------------------------
// Attention logits s,d per (node,head): 128-length dot with att vectors.
// ---------------------------------------------------------------------------
__global__ void attn_sd_kernel(const float* __restrict__ XP,
                               const float* __restrict__ asrc,
                               const float* __restrict__ adst,
                               float* __restrict__ S, float* __restrict__ D, int N) {
    const int idx = blockIdx.x * blockDim.x + threadIdx.x;
    if (idx >= N * HEADS) return;
    const int n = idx >> 2, h = idx & 3;
    const float* xr = XP + (size_t)n * 512 + h * 128;
    const float* as = asrc + h * 128;
    const float* ad = adst + h * 128;
    float s = 0.f, d = 0.f;
#pragma unroll 4
    for (int c = 0; c < 128; ++c) { const float x = xr[c]; s += x * as[c]; d += x * ad[c]; }
    S[idx] = s; D[idx] = d;
}

__global__ void init_md_kernel(float* __restrict__ M, float* __restrict__ den, int n4) {
    const int i = blockIdx.x * blockDim.x + threadIdx.x;
    if (i < n4) { M[i] = -1e30f; den[i] = 0.f; }
}

__global__ void zero_f_kernel(float* __restrict__ p, long long n) {
    const long long i = (long long)blockIdx.x * blockDim.x + threadIdx.x;
    if (i < n) p[i] = 0.f;
}

__device__ inline void atomicMaxFloat(float* addr, float v) {
    if (v >= 0.f) atomicMax((int*)addr, __float_as_int(v));
    else          atomicMin((unsigned int*)addr, __float_as_uint(v));
}

__device__ inline float lrelu(float x) { return (x > 0.f) ? x : x * NEG_SLOPE; }

// segment max over dst; also pre-warm L2 with the xp[src] rows that the
// scatter kernel will gather (global_prefetch_b8; 61MB xp fits 192MB L2).
__global__ void edge_max_kernel(const int* __restrict__ ei, const float* __restrict__ S,
                                const float* __restrict__ D, const float* __restrict__ XP,
                                float* __restrict__ M, int E) {
    const int i = blockIdx.x * blockDim.x + threadIdx.x;
    if (i >= E * HEADS) return;
    const int e = i >> 2, h = i & 3;
    const int src = ei[e], dst = ei[E + e];
    if (h == 0) {
        __builtin_prefetch(XP + (size_t)src * 512, 0, 1);        // head 0/1 halves
        __builtin_prefetch(XP + (size_t)src * 512 + 256, 0, 1);  // head 2/3 halves
    }
    atomicMaxFloat(&M[dst * 4 + h], lrelu(S[src * 4 + h] + D[dst * 4 + h]));
}

// exp(e - m[dst]) and segment-sum denom
__global__ void edge_exp_kernel(const int* __restrict__ ei, const float* __restrict__ S,
                                const float* __restrict__ D, const float* __restrict__ M,
                                float* __restrict__ EE, float* __restrict__ den, int E) {
    const int i = blockIdx.x * blockDim.x + threadIdx.x;
    if (i >= E * HEADS) return;
    const int e = i >> 2, h = i & 3;
    const int src = ei[e], dst = ei[E + e];
    const float ev = __expf(lrelu(S[src * 4 + h] + D[dst * 4 + h]) - M[dst * 4 + h]);
    EE[i] = ev;
    atomicAdd(&den[dst * 4 + h], ev);
}

// weighted scatter: agg[dst,h,:] += alpha * xp[src,h,:], one block per edge,
// thread t covers channels 4t..4t+3 (stays within one head since 4|128).
__global__ __launch_bounds__(128)
void edge_scatter_kernel(const int* __restrict__ ei, const float* __restrict__ XP,
                         const float* __restrict__ EE, const float* __restrict__ den,
                         float* __restrict__ AGG, int E) {
    const int e = blockIdx.x;
    if (e >= E) return;
    const int t = threadIdx.x;
    const int src = ei[e], dst = ei[E + e];
    const int idx0 = t * 4;
    const int h = idx0 >> 7;
    const float alpha = EE[e * 4 + h] / (den[dst * 4 + h] + 1e-16f);
    const float* xs = XP + (size_t)src * 512 + idx0;
    float* ag = AGG + (size_t)dst * 512 + idx0;
#pragma unroll
    for (int j = 0; j < 4; ++j) atomicAdd(ag + j, xs[j] * alpha);
}

// head-mean + conv bias + residual + lin bias, ReLU, JK accumulate.
__global__ void combine_kernel(const float* __restrict__ AGG, const float* __restrict__ cb,
                               const float* __restrict__ R, const float* __restrict__ lb,
                               float* __restrict__ Xout, float* __restrict__ XF,
                               int N, int initXF) {
    const int i = blockIdx.x * blockDim.x + threadIdx.x;
    if (i >= N * 128) return;
    const int n = i >> 7, c = i & 127;
    const float* a = AGG + (size_t)n * 512 + c;
    float v = (a[0] + a[128] + a[256] + a[384]) * 0.25f + cb[c] + R[i] + lb[c];
    v = v > 0.f ? v : 0.f;
    Xout[i] = v;
    XF[i] = initXF ? v : (XF[i] + v);
}

__global__ void f32_to_bf16_kernel(const float* __restrict__ X, __bf16* __restrict__ Y,
                                   long long n) {
    const long long i = (long long)blockIdx.x * blockDim.x + threadIdx.x;
    if (i < n) Y[i] = (__bf16)X[i];
}

// ---------------------------------------------------------------------------
// Host launcher
// ---------------------------------------------------------------------------
extern "C" void kernel_launch(void* const* d_in, const int* in_sizes, int n_in,
                              void* d_out, int out_size, void* d_ws, size_t ws_size,
                              hipStream_t stream) {
    const float* x        = (const float*)d_in[0];
    const int*   ei       = (const int*)d_in[1];     // edge_index [2,E]
    const float* ln_scale = (const float*)d_in[2];   // [2,128]
    const float* ln_bias  = (const float*)d_in[3];
    const float* convW    = (const float*)d_in[4];   // [2,128,512]
    const float* att_src  = (const float*)d_in[5];   // [2,4,128]
    const float* att_dst  = (const float*)d_in[6];
    const float* conv_b   = (const float*)d_in[7];   // [2,128]
    const float* linW     = (const float*)d_in[8];   // [2,128,128]
    const float* lin_b    = (const float*)d_in[9];
    const float* predW    = (const float*)d_in[10];  // [128,64]
    const float* pred_b   = (const float*)d_in[11];  // [64]
    float* out = (float*)d_out;

    const int N = in_sizes[0] / 128;
    const int E = in_sizes[1] / 2;

    // Workspace carve-out (256B aligned slices)
    char* ws = (char*)d_ws;
    auto alloc = [&](size_t bytes) -> char* {
        char* p = ws;
        ws += (bytes + 255) & ~(size_t)255;
        return p;
    };
    __bf16* h_bf   = (__bf16*)alloc((size_t)N * 128 * sizeof(__bf16));
    float*  xp     = (float*) alloc((size_t)N * 512 * sizeof(float));
    float*  resid  = (float*) alloc((size_t)N * 128 * sizeof(float));
    float*  Sv     = (float*) alloc((size_t)N * 4 * sizeof(float));
    float*  Dv     = (float*) alloc((size_t)N * 4 * sizeof(float));
    float*  Mv     = (float*) alloc((size_t)N * 4 * sizeof(float));
    float*  denom  = (float*) alloc((size_t)N * 4 * sizeof(float));
    float*  eexp   = (float*) alloc((size_t)E * 4 * sizeof(float));
    float*  agg    = (float*) alloc((size_t)N * 512 * sizeof(float));
    float*  xbuf   = (float*) alloc((size_t)N * 128 * sizeof(float));
    float*  xfinal = (float*) alloc((size_t)N * 128 * sizeof(float));
    __bf16* xf_bf  = (__bf16*)alloc((size_t)N * 128 * sizeof(__bf16));
    __bf16* Bconv  = (__bf16*)alloc((size_t)128 * 512 * sizeof(__bf16));
    __bf16* Blin   = (__bf16*)alloc((size_t)128 * 128 * sizeof(__bf16));
    __bf16* Bpred  = (__bf16*)alloc((size_t)128 * 64 * sizeof(__bf16));
    (void)ws_size;

    const int mtiles = (N + 15) >> 4;
    auto gemmBlocks = [&](int P) { return (mtiles * (P >> 6) + 3) / 4; };

    for (int layer = 0; layer < 2; ++layer) {
        const float* xin = (layer == 0) ? x : xbuf;

        // Pack weights into WMMA B-fragment order (bf16).
        pack_b_kernel<<<(128 * 512 + 255) / 256, 256, 0, stream>>>(
            convW + (size_t)layer * 128 * 512, Bconv, 128, 512);
        pack_b_kernel<<<(128 * 128 + 255) / 256, 256, 0, stream>>>(
            linW + (size_t)layer * 128 * 128, Blin, 128, 128);

        // LayerNorm -> bf16 h
        ln_bf16_kernel<<<N, 128, 0, stream>>>(
            xin, ln_scale + layer * 128, ln_bias + layer * 128, h_bf, N);

        // xp = h @ convW  (N x 512), resid = h @ linW (N x 128)  — WMMA
        gemm_wmma_kernel<<<gemmBlocks(512), 128, 0, stream>>>(
            h_bf, Bconv, nullptr, xp, N, 512, 0);
        gemm_wmma_kernel<<<gemmBlocks(128), 128, 0, stream>>>(
            h_bf, Blin, nullptr, resid, N, 128, 0);

        // Attention logits
        attn_sd_kernel<<<(N * HEADS + 255) / 256, 256, 0, stream>>>(
            xp, att_src + (size_t)layer * 4 * 128, att_dst + (size_t)layer * 4 * 128,
            Sv, Dv, N);

        // Segment softmax + aggregation
        init_md_kernel<<<(N * 4 + 255) / 256, 256, 0, stream>>>(Mv, denom, N * 4);
        zero_f_kernel<<<(int)(((long long)N * 512 + 255) / 256), 256, 0, stream>>>(
            agg, (long long)N * 512);
        edge_max_kernel<<<(E * HEADS + 255) / 256, 256, 0, stream>>>(
            ei, Sv, Dv, xp, Mv, E);
        edge_exp_kernel<<<(E * HEADS + 255) / 256, 256, 0, stream>>>(
            ei, Sv, Dv, Mv, eexp, denom, E);
        edge_scatter_kernel<<<E, 128, 0, stream>>>(ei, xp, eexp, denom, agg, E);

        // conv + residual + relu, JK sum
        combine_kernel<<<(N * 128 + 255) / 256, 256, 0, stream>>>(
            agg, conv_b + layer * 128, resid, lin_b + layer * 128,
            xbuf, xfinal, N, layer == 0 ? 1 : 0);
    }

    // Final projection: out = xfinal @ predW + pred_b  (WMMA)
    f32_to_bf16_kernel<<<(int)(((long long)N * 128 + 255) / 256), 256, 0, stream>>>(
        xfinal, xf_bf, (long long)N * 128);
    pack_b_kernel<<<(128 * 64 + 255) / 256, 256, 0, stream>>>(predW, Bpred, 128, 64);
    gemm_wmma_kernel<<<gemmBlocks(64), 128, 0, stream>>>(
        xf_bf, Bpred, pred_b, out, N, 64, 1);
}
